// ScaledDotProductAttention_27616639713416
// MI455X (gfx1250) — compile-verified, matrix operand order checked by reading, not run on hardware
//
#include <hip/hip_runtime.h>

// ScaledDotProductAttention, faithful to the reference source: the two
// jnp.where() calls overwrite *every* score, so attn depends only on attn_mask
// (uniform 1/n_unmasked over unmasked keys; 1/S if a row is fully masked) and
// context = attn @ V.  Q and K are dead inputs.
//
// Roofline: 537 MB attn store (non-temporal) + 134 MB mask read (non-temporal)
// ~= 30 us @ 23.3 TB/s.  The only real math is attn@V (17.2 GFLOP) on
// v_wmma_f32_16x16x32_f16, with V pre-swizzled once to f16 fragment order in
// d_ws (8 MB, L2-resident, regular-temporal) so each B fragment is one
// contiguous 32 B/lane load and each A fragment needs two ds_load_b64.

typedef __attribute__((ext_vector_type(16))) _Float16     v16h;
typedef __attribute__((ext_vector_type(2)))  _Float16     v2h;
typedef __attribute__((ext_vector_type(8)))  float        v8f;
typedef __attribute__((ext_vector_type(4)))  unsigned int u32x4;
typedef __attribute__((ext_vector_type(4)))  float        f32x4;
typedef __attribute__((ext_vector_type(2)))  unsigned int u32x2;

#define B_    2
#define H_    16
#define S_    2048
#define D_    64
#define QT    16          // query rows per workgroup
#define NWAVE 8           // 256 threads, wave32
#define NCHNK (S_ / 32)   // 64 key chunks of K=32
#define CPW   (NCHNK / NWAVE)  // 8 chunks per wave

// ---------------------------------------------------------------------------
// Pre-swizzle V (f32, [BH,S,D]) into f16 WMMA-B fragment order:
//   Wv[(((bh*64 + c)*4 + s)*32 + lane)*16 + e]
//     = (f16) V[bh][c*32 + K(lane,e)][s*16 + (lane&15)]
// with K(lane,e) = (j>=4?16:0) + ((lane>>4)<<3) + 2*(j&3) + (e&1),  j = e>>1.
// Each thread produces one packed f16 pair (e = 2j, 2j+1 -> K, K+1).
// ---------------------------------------------------------------------------
__global__ __launch_bounds__(256) void pack_v_kernel(
    const float* __restrict__ V, _Float16* __restrict__ Wv)
{
    int p    = blockIdx.x * 256 + threadIdx.x;   // pair index
    int j    = p & 7;
    int lane = (p >> 3) & 31;
    int s    = (p >> 8) & 3;
    int c    = (p >> 10) & 63;
    int bh   = p >> 16;

    int kg = (lane >> 4) << 3;
    int k  = c * 32 + ((j >= 4) ? 16 : 0) + kg + ((j & 3) << 1);
    int d  = (s << 4) + (lane & 15);

    size_t src = ((size_t)bh * S_ + k) * D_ + d;
    float f0 = __builtin_nontemporal_load(V + src);       // V f32 dead after this
    float f1 = __builtin_nontemporal_load(V + src + D_);  // K+1 -> +D floats
    v2h hp;
    hp.x = (_Float16)f0;
    hp.y = (_Float16)f1;
    ((v2h*)Wv)[p] = hp;                                   // dest coalesced, RT
}

// ---------------------------------------------------------------------------
// Main kernel.  USE_WS selects the pre-swizzled f16 V path.
// ---------------------------------------------------------------------------
template <bool USE_WS>
__global__ __launch_bounds__(256) void sdpa_mask_wmma_kernel(
    const unsigned char* __restrict__ mask,   // [B,H,S,S] bool (1 byte, 0/1)
    const float*         __restrict__ V,      // [B,H,S,D] f32 (fallback path)
    const _Float16*      __restrict__ Wv,     // pre-swizzled f16 V (fast path)
    float* __restrict__ ctx_out,              // [B,H,S,D] f32
    float* __restrict__ attn_out)             // [B,H,S,S] f32
{
    __shared__ __align__(16) unsigned char smask[QT * S_];  // 32 KB mask tile
    __shared__ float        sred[QT * D_];                  // 4 KB ctx reduction
    __shared__ unsigned int scnt[QT];
    __shared__ float        pU[QT], pM[QT];

    const int tid  = threadIdx.x;
    const int wave = tid >> 5;
    const int lane = tid & 31;

    const int blk = blockIdx.x;
    const int bh  = blk >> 7;          // S_/QT = 128 q-tiles per (b,h)
    const int qt  = blk & 127;
    const int q0  = qt * QT;

    const size_t mask_base = ((size_t)bh * S_ + q0) * S_;
    const size_t v_base    = (size_t)bh * S_ * D_;
    const size_t ctx_base  = ((size_t)bh * S_ + q0) * D_;
    const size_t attn_base = ((size_t)bh * S_ + q0) * S_;

    // ---- init shared counters / reduction buffer ----
    if (tid < QT) scnt[tid] = 0u;
    #pragma unroll
    for (int i = 0; i < (QT * D_) / 256; ++i) sred[i * 256 + tid] = 0.0f;
    __syncthreads();

    // ---- Phase 1: stage mask tile to LDS (NT loads), count masked per row ----
    {
        const u32x4* gm = (const u32x4*)(mask + mask_base);
        u32x4*       sm = (u32x4*)smask;
        #pragma unroll
        for (int i = 0; i < 8; ++i) {
            int   idx = i * 256 + tid;         // 2048 16-byte groups, 128 per row
            u32x4 g   = __builtin_nontemporal_load(&gm[idx]);  // read-once
            sm[idx]   = g;
            // bool bytes are 0/1, so popcount == byte sum
            unsigned cnt = (unsigned)(__popc(g.x) + __popc(g.y) +
                                      __popc(g.z) + __popc(g.w));
            atomicAdd(&scnt[idx >> 7], cnt);
        }
    }
    __syncthreads();

    if (tid < QT) {
        unsigned masked = scnt[tid];
        unsigned un     = (unsigned)S_ - masked;
        if (un > 0u) { pU[tid] = 1.0f / (float)un; pM[tid] = 0.0f; }
        else         { pU[tid] = 1.0f / (float)S_; pM[tid] = 1.0f / (float)S_; }
    }
    __syncthreads();

    // ---- Phase 2a: coalesced attn tile store (dominant HBM traffic, NT) ----
    {
        const unsigned* sm32 = (const unsigned*)smask;
        #pragma unroll 4
        for (int i = 0; i < 32; ++i) {
            int idx4 = i * 256 + tid;          // 8192 groups of 4 bytes
            int row  = idx4 >> 9;              // 512 groups per row
            int col  = (idx4 & 511) << 2;
            unsigned u = sm32[idx4];
            float u0 = pU[row], m0 = pM[row];
            f32x4 f;
            f.x = (u & 0x000000FFu) ? m0 : u0;
            f.y = (u & 0x0000FF00u) ? m0 : u0;
            f.z = (u & 0x00FF0000u) ? m0 : u0;
            f.w = (u & 0xFF000000u) ? m0 : u0;
            // write-once stream: keep it out of L2's way
            __builtin_nontemporal_store(
                f, (f32x4*)(attn_out + attn_base + (size_t)row * S_ + col));
        }
    }

    // ---- Phase 2b: context = attn_tile @ V via v_wmma_f32_16x16x32_f16 ----
    // A (16x32 f16, probs): lane&15 = row M; lanes 0-15 hold K in {0..7,16..23},
    // lanes 16-31 hold K in {8..15,24..31}: two contiguous 8-byte mask runs
    // -> two ds_load_b64 + bit tests.
    // B (32x16 f16, V slice): pre-swizzled fragment (32 B/lane contiguous).
    v8f acc[4] = {v8f{}, v8f{}, v8f{}, v8f{}};
    {
        const int m  = lane & 15;              // query row (A) / d column (B)
        const int kg = (lane >> 4) << 3;       // 0 or 8
        const _Float16 hU = (_Float16)pU[m];
        const _Float16 hM = (_Float16)pM[m];
        const _Float16* wvb = Wv + (size_t)bh * (S_ * D_);   // fragment blocks
        const float*    Vb  = V + v_base;

        #pragma unroll
        for (int it = 0; it < CPW; ++it) {     // static trip count: no exec loop
            const int c   = it * NWAVE + wave;
            const int kc0 = c * 32;

            // A fragment from two 8-byte LDS loads
            v16h a;
            {
                const unsigned char* mrow = smask + m * S_ + kc0 + kg;
                u32x2 lo = *(const u32x2*)(mrow);        // K = kg .. kg+7
                u32x2 hi = *(const u32x2*)(mrow + 16);   // K = kg+16 .. kg+23
                unsigned w0 = lo.x, w1 = lo.y, w2 = hi.x, w3 = hi.y;
                #pragma unroll
                for (int e = 0; e < 4; ++e) {
                    a[e]      = (w0 & (0xFFu << (8 * e))) ? hM : hU;
                    a[e + 4]  = (w1 & (0xFFu << (8 * e))) ? hM : hU;
                    a[e + 8]  = (w2 & (0xFFu << (8 * e))) ? hM : hU;
                    a[e + 12] = (w3 & (0xFFu << (8 * e))) ? hM : hU;
                }
            }

            // Preload all four B fragments (8 outstanding b128 loads), then
            // issue the 4 WMMAs back to back.
            v16h b[4];
            #pragma unroll
            for (int s = 0; s < 4; ++s) {
                if (USE_WS) {
                    b[s] = *(const v16h*)(wvb + (((size_t)(c * 4 + s)) << 9)
                                              + (lane << 4));
                } else {
                    #pragma unroll
                    for (int e = 0; e < 16; ++e) {
                        int j = e >> 1;
                        int k = ((j >= 4) ? 16 : 0) + kg + ((j & 3) << 1) + (e & 1);
                        b[s][e] = (_Float16)Vb[(size_t)(kc0 + k) * D_ + (s << 4) + m];
                    }
                }
            }
            #pragma unroll
            for (int s = 0; s < 4; ++s) {
                acc[s] = __builtin_amdgcn_wmma_f32_16x16x32_f16(
                    false, a, false, b[s], (short)0, acc[s], false, false);
            }
        }
    }

    // ---- Phase 3: cross-wave reduction (ds_add_f32) + coalesced ctx store ----
    {
        const int mg = (lane >> 4) << 3;       // C/D layout: M = mg + vgpr
        #pragma unroll
        for (int s = 0; s < 4; ++s) {
            #pragma unroll
            for (int v = 0; v < 8; ++v) {
                int mm = mg + v;
                int d  = (s << 4) + (lane & 15);
                atomicAdd(&sred[mm * D_ + d], acc[s][v]);
            }
        }
    }
    __syncthreads();

    #pragma unroll
    for (int i = 0; i < (QT * D_) / 256; ++i) {
        int idx = i * 256 + tid;               // idx = m*64 + d
        ctx_out[ctx_base + idx] = sred[idx];
    }
}

extern "C" void kernel_launch(void* const* d_in, const int* in_sizes, int n_in,
                              void* d_out, int out_size, void* d_ws, size_t ws_size,
                              hipStream_t stream) {
    // setup_inputs order: Q, K, V, attn_mask.  Q and K are dead (see header).
    const float*         V    = (const float*)d_in[2];
    const unsigned char* mask = (const unsigned char*)d_in[3];  // jax bool = u8

    float* ctx  = (float*)d_out;                                   // [B,H,S,D]
    float* attn = (float*)d_out + (size_t)B_ * H_ * S_ * D_;       // [B,H,S,S]

    const size_t wv_bytes = (size_t)B_ * H_ * S_ * D_ * sizeof(_Float16); // 8 MB
    dim3 grid(B_ * H_ * (S_ / QT));   // 4096 workgroups
    dim3 block(256);                   // 8 waves (wave32)

    if (ws_size >= wv_bytes) {
        _Float16* Wv = (_Float16*)d_ws;
        int npair = (B_ * H_ * S_ * D_) / 2;
        pack_v_kernel<<<dim3(npair / 256), block, 0, stream>>>(V, Wv);
        sdpa_mask_wmma_kernel<true><<<grid, block, 0, stream>>>(mask, V, Wv, ctx, attn);
    } else {
        sdpa_mask_wmma_kernel<false><<<grid, block, 0, stream>>>(mask, V, nullptr, ctx, attn);
    }
}
